// RNN_50783693308479
// MI455X (gfx1250) — compile-verified, matrix operand order checked by reading, not run on hardware
//
#include <hip/hip_runtime.h>
#include <hip/hip_bf16.h>
#include <math.h>

// ---------------------------------------------------------------------------
// RNN on MI455X (gfx1250, wave32, WMMA).
//   h_{t+1} = tanh([x_t | h_t] @ [W_xh ; W_hh] + b_h)   (fused K = 512+1024)
//   y       = h_S @ W_yh + b_y
// bf16 WMMA (v_wmma_f32_16x16x32_bf16) with fp32 accumulation.
// Persistent kernel, device-scope barrier per step. All operands pre-swizzled
// into the ISA fragment layout so the hot loop is b128 loads + v_wmma only.
// ---------------------------------------------------------------------------

typedef __attribute__((ext_vector_type(16))) __bf16 v16bf;
typedef __attribute__((ext_vector_type(8)))  float  v8f;

#define BATCH 128
#define SEQ   512
#define ISZ   512
#define HSZ   1024
#define OSZ   512

#define MT     (BATCH/16)      // 8  row tiles of h / y
#define NT_H   (HSZ/16)        // 64 col tiles of h
#define NT_O   (OSZ/16)        // 32 col tiles of y
#define KT_X   (ISZ/32)        // 16 k-tiles from x
#define KT_H   (HSZ/32)        // 32 k-tiles from h
#define KT_ALL (KT_X + KT_H)   // 48 combined k-tiles

#define FRAG_ELEMS 512         // one 16x32 (or 32x16) bf16 tile = 32 lanes * 16

// Workspace layout (bytes)
#define CTR_OFF  ((size_t)0)
#define W_OFF    ((size_t)256)
#define W_BYTES  ((size_t)KT_ALL * NT_H * FRAG_ELEMS * 2)   // 3 MiB  [Wxh;Whh]
#define Y_OFF    (W_OFF + W_BYTES)
#define Y_BYTES  ((size_t)KT_H * NT_O * FRAG_ELEMS * 2)     // 1 MiB  Wyh
#define H_OFF    (Y_OFF + Y_BYTES)
#define H_BYTES  ((size_t)MT * KT_H * FRAG_ELEMS * 2)       // 256 KiB per h buf
#define X_OFF    (H_OFF + 2 * H_BYTES)
#define X_STEP_ELEMS ((size_t)MT * KT_X * FRAG_ELEMS)       // 65536 elems / step

// fp32 -> bf16, round-to-nearest-even
__device__ __forceinline__ __bf16 f2bf(float f) {
    union { float f; unsigned u; } v; v.f = f;
    unsigned r = v.u + 0x7FFFu + ((v.u >> 16) & 1u);
    union { unsigned short s; __bf16 b; } o;
    o.s = (unsigned short)(r >> 16);
    return o.b;
}

// ISA 7.12.2: 16-bit A-matrix 16x32 fragment position for element (Mn, K).
// half = (K%16)/8 ; VGPR j = 4*(K/16) + (K%8)/2 ; pair = K%2
// lane = Mn + 16*half ; slot-in-lane = 2*j + pair.  (B 32x16 uses N for Mn.)
__device__ __forceinline__ void frag_pos(int Mn, int K, int& lane, int& slot) {
    int half = (K & 15) >> 3;
    int j    = ((K >> 4) << 2) + ((K & 7) >> 1);
    lane = Mn + (half << 4);
    slot = (j << 1) | (K & 1);
}

// --------------------------- init: zero h0 + counter ------------------------
__global__ __launch_bounds__(256) void rnn_init(__bf16* hfrag, unsigned* ctr) {
    size_t i = (size_t)blockIdx.x * 256 + threadIdx.x;
    size_t n = 2 * (H_BYTES / 2);
    if (i < n) hfrag[i] = f2bf(0.0f);
    if (blockIdx.x == 0 && threadIdx.x == 0) *ctr = 0u;
}

// ---------------- pack [W_xh ; W_hh] (1536 x 1024) into B-fragments ---------
__global__ __launch_bounds__(256) void pack_w(const float* __restrict__ Wxh,
                                              const float* __restrict__ Whh,
                                              __bf16* __restrict__ wfrag) {
    int idx = blockIdx.x * 256 + threadIdx.x;       // 1536*1024 = 1572864
    if (idx >= (ISZ + HSZ) * HSZ) return;
    int k = idx >> 10;                              // combined K row
    int n = idx & (HSZ - 1);                        // hidden column
    float v = (k < ISZ) ? Wxh[(size_t)k * HSZ + n]
                        : Whh[(size_t)(k - ISZ) * HSZ + n];
    int kt = k >> 5, K = k & 31, nt = n >> 4, N = n & 15;
    int lane, slot; frag_pos(N, K, lane, slot);
    wfrag[((size_t)(kt * NT_H + nt)) * FRAG_ELEMS + lane * 16 + slot] = f2bf(v);
}

// ---------------- pack W_yh (1024 x 512) into B-fragments -------------------
__global__ __launch_bounds__(256) void pack_y(const float* __restrict__ Wyh,
                                              __bf16* __restrict__ yfrag) {
    int idx = blockIdx.x * 256 + threadIdx.x;       // 1024*512 = 524288
    if (idx >= HSZ * OSZ) return;
    int k = idx / OSZ, n = idx % OSZ;
    int kt = k >> 5, K = k & 31, nt = n >> 4, N = n & 15;
    int lane, slot; frag_pos(N, K, lane, slot);
    yfrag[((size_t)(kt * NT_O + nt)) * FRAG_ELEMS + lane * 16 + slot] =
        f2bf(Wyh[idx]);
}

// ---------------- pack x [B,S,I] fp32 into per-step A-fragments -------------
__global__ __launch_bounds__(256) void pack_x(const float* __restrict__ x,
                                              __bf16* __restrict__ xfrag) {
    size_t idx = (size_t)blockIdx.x * 256 + threadIdx.x;  // 128*512*512
    if (idx >= (size_t)BATCH * SEQ * ISZ) return;
    int b = (int)(idx >> 18);          // / (512*512)
    int r = (int)(idx & 262143);
    int t = r >> 9;                    // / 512
    int i = r & 511;
    int mt = b >> 4, M = b & 15, kt = i >> 5, K = i & 31;
    int lane, slot; frag_pos(M, K, lane, slot);
    xfrag[(size_t)t * X_STEP_ELEMS +
          ((size_t)(mt * KT_X + kt)) * FRAG_ELEMS + lane * 16 + slot] =
        f2bf(x[idx]);
}

// --------------------------- persistent recurrence --------------------------
// 128 blocks x 128 threads (4 wave32s) = 512 waves; wave w owns output tile
// (mt = w/64, nt = w%64) of h[128x1024]. After the scan, waves 0..255 do the
// 128x512 output GEMM.
__global__ __launch_bounds__(128) void rnn_persist(
    const __bf16* __restrict__ xfrag, const __bf16* __restrict__ wfrag,
    const __bf16* __restrict__ yfrag, __bf16* __restrict__ hA0,
    __bf16* __restrict__ hA1, const float* __restrict__ bh,
    const float* __restrict__ by, float* __restrict__ out,
    unsigned* __restrict__ ctr) {

    const int lane = threadIdx.x & 31;
    const int wg   = blockIdx.x * 4 + (threadIdx.x >> 5);   // 0..511
    const int mt   = wg >> 6;                                // 0..7
    const int nt   = wg & 63;                                // 0..63
    const unsigned nblocks = gridDim.x;

    // C/D layout: lane -> (M = v + 8*(lane/16), N = lane%16)
    const int N    = lane & 15;
    const int rsel = lane >> 4;

    // Precomputed scatter map: h_new(mt*16+M, nt*16+N) -> A-fragment slot for
    // tile (mt, nt/2), K = (nt%2)*16 + N. half/j/slot depend only on N,nt.
    const int Kst   = ((nt & 1) << 4) | N;
    const int halfK = (Kst & 15) >> 3;
    const int jK    = ((Kst >> 4) << 2) + ((Kst & 7) >> 1);
    const int slotK = (jK << 1) | (Kst & 1);
    const size_t hdstBase = ((size_t)(mt * KT_H + (nt >> 1))) * FRAG_ELEMS;

    const float bhv = bh[nt * 16 + N];

    for (int t = 0; t < SEQ; ++t) {
        const __bf16* hcur = (t & 1) ? hA1 : hA0;
        __bf16*       hnxt = (t & 1) ? hA0 : hA1;

        v8f acc = {};
        const __bf16* xa = xfrag + (size_t)t * X_STEP_ELEMS +
                           ((size_t)mt * KT_X) * FRAG_ELEMS;
        if (t + 1 < SEQ)  // warm L2 for next step's x tiles
            __builtin_prefetch(xa + X_STEP_ELEMS, 0, 1);

        // K = 0..511 : x_t @ W_xh
        #pragma unroll 4
        for (int kt = 0; kt < KT_X; ++kt) {
            v16bf a = *(const v16bf*)(xa + (size_t)kt * FRAG_ELEMS + lane * 16);
            v16bf b = *(const v16bf*)(wfrag +
                       ((size_t)(kt * NT_H + nt)) * FRAG_ELEMS + lane * 16);
            acc = __builtin_amdgcn_wmma_f32_16x16x32_bf16(
                false, a, false, b, (short)0, acc, false, false);
        }
        // K = 512..1535 : h_t @ W_hh
        #pragma unroll 4
        for (int kt = 0; kt < KT_H; ++kt) {
            v16bf a = *(const v16bf*)(hcur +
                       ((size_t)(mt * KT_H + kt)) * FRAG_ELEMS + lane * 16);
            v16bf b = *(const v16bf*)(wfrag +
                       ((size_t)((KT_X + kt) * NT_H + nt)) * FRAG_ELEMS +
                       lane * 16);
            acc = __builtin_amdgcn_wmma_f32_16x16x32_bf16(
                false, a, false, b, (short)0, acc, false, false);
        }

        // tanh + scatter h_new directly into next step's A-fragment layout
        #pragma unroll
        for (int v = 0; v < 8; ++v) {
            float hv = tanhf(acc[v] + bhv);
            int lanep = (v + 8 * rsel) + (halfK << 4);
            hnxt[hdstBase + lanep * 16 + slotK] = f2bf(hv);
        }

        // ---- device-wide barrier (monotonic counter, release/acquire) ----
        __threadfence();
        __syncthreads();
        if (threadIdx.x == 0) {
            __hip_atomic_fetch_add(ctr, 1u, __ATOMIC_RELEASE,
                                   __HIP_MEMORY_SCOPE_AGENT);
            unsigned target = (unsigned)(t + 1) * nblocks;
            while (__hip_atomic_load(ctr, __ATOMIC_ACQUIRE,
                                     __HIP_MEMORY_SCOPE_AGENT) < target)
                __builtin_amdgcn_s_sleep(2);
        }
        __syncthreads();
    }

    // -------- output GEMM: y = h_S @ W_yh + b_y  (SEQ even -> h_S in hA0) ---
    if (wg < MT * NT_O) {                       // 256 tiles
        const int omt = wg >> 5, ont = wg & 31;
        v8f acc = {};
        #pragma unroll 4
        for (int kt = 0; kt < KT_H; ++kt) {
            v16bf a = *(const v16bf*)(hA0 +
                       ((size_t)(omt * KT_H + kt)) * FRAG_ELEMS + lane * 16);
            v16bf b = *(const v16bf*)(yfrag +
                       ((size_t)(kt * NT_O + ont)) * FRAG_ELEMS + lane * 16);
            acc = __builtin_amdgcn_wmma_f32_16x16x32_bf16(
                false, a, false, b, (short)0, acc, false, false);
        }
        const float byv = by[ont * 16 + N];
        #pragma unroll
        for (int v = 0; v < 8; ++v) {
            int M = v + 8 * rsel;
            out[(size_t)(omt * 16 + M) * OSZ + ont * 16 + N] = acc[v] + byv;
        }
    }
}

// ---------------------------------------------------------------------------
extern "C" void kernel_launch(void* const* d_in, const int* in_sizes, int n_in,
                              void* d_out, int out_size, void* d_ws,
                              size_t ws_size, hipStream_t stream) {
    const float* x   = (const float*)d_in[0];
    const float* Wxh = (const float*)d_in[1];
    const float* Whh = (const float*)d_in[2];
    const float* Wyh = (const float*)d_in[3];
    const float* bh  = (const float*)d_in[4];
    const float* by  = (const float*)d_in[5];
    float* out = (float*)d_out;

    char* ws = (char*)d_ws;
    unsigned* ctr  = (unsigned*)(ws + CTR_OFF);
    __bf16* wfrag  = (__bf16*)(ws + W_OFF);
    __bf16* yfrag  = (__bf16*)(ws + Y_OFF);
    __bf16* hA0    = (__bf16*)(ws + H_OFF);
    __bf16* hA1    = (__bf16*)(ws + H_OFF + H_BYTES);
    __bf16* xfrag  = (__bf16*)(ws + X_OFF);

    // re-init every call: h0 = 0, barrier counter = 0 (graph-replay safe)
    rnn_init<<<(unsigned)((2 * (H_BYTES / 2) + 255) / 256), 256, 0, stream>>>(
        hA0, ctr);
    pack_w<<<((ISZ + HSZ) * HSZ) / 256, 256, 0, stream>>>(Wxh, Whh, wfrag);
    pack_y<<<(HSZ * OSZ) / 256, 256, 0, stream>>>(Wyh, yfrag);
    pack_x<<<(unsigned)(((size_t)BATCH * SEQ * ISZ) / 256), 256, 0, stream>>>(
        x, xfrag);
    rnn_persist<<<128, 128, 0, stream>>>(xfrag, wfrag, yfrag, hA0, hA1, bh, by,
                                         out, ctr);
}
